// VectorQuantizer_52656299049012
// MI455X (gfx1250) — matrix operand order repaired and checked by reading, not compiled, not run on hardware
//
#include <hip/hip_runtime.h>
#include <hip/hip_bf16.h>

typedef __attribute__((ext_vector_type(16))) __bf16 v16bf;
typedef __attribute__((ext_vector_type(8)))  __bf16 v8bf;
typedef __attribute__((ext_vector_type(8)))  float  v8f;

#define D_DIM 256
#define K_CODES 1024
#define M_TILE 32   // rows per block (2 WMMA row-tiles per wave, shared B)

// ---------------------------------------------------------------------------
// Kernel 0: transpose codebook (f32 + bf16), precompute ||e||^2, zero loss.
// embeddings: [D=256][K=1024] f32.  One block per code k, 256 threads over d.
// ---------------------------------------------------------------------------
__global__ void vq_prep(const float* __restrict__ emb,
                        float* __restrict__ eTf,    // [K][D] f32
                        __bf16* __restrict__ eTb,   // [K][D] bf16
                        float* __restrict__ enorm,  // [K]
                        float* __restrict__ lossptr) {
  const int k = blockIdx.x;
  const int d = threadIdx.x;
  float v = emb[d * K_CODES + k];
  eTf[k * D_DIM + d] = v;
  eTb[k * D_DIM + d] = (__bf16)v;
  __shared__ float red[256];
  red[d] = v * v;
  __syncthreads();
  for (int s = 128; s > 0; s >>= 1) {
    if (d < s) red[d] += red[d + s];
    __syncthreads();
  }
  if (d == 0) enorm[k] = red[0];
  if (k == 0 && d == 0) *lossptr = 0.0f;
}

// ---------------------------------------------------------------------------
// Kernel 1: per block: 32 input rows vs all 1024 codes.
// 256 threads = 8 waves; wave w handles code tiles ct = ct8*8 + w (16 codes),
// and TWO 16-row tiles sharing each B fragment (halves L2 B-traffic, and the
// two independent accumulator chains cover WMMA->WMMA RAW latency).
// ---------------------------------------------------------------------------
__global__ void __launch_bounds__(256)
vq_main(const float* __restrict__ x,
        const float* __restrict__ eTf,
        const __bf16* __restrict__ eTb,
        const float* __restrict__ enorm,
        float* __restrict__ out,
        float* __restrict__ lossptr,
        float loss_scale) {
  __shared__ __align__(16) float  sAf[M_TILE][D_DIM];   // 32 KB
  __shared__ __align__(16) __bf16 sAb[M_TILE][D_DIM];   // 16 KB
  __shared__ float sXn[M_TILE];
  __shared__ float sEn[K_CODES];                        // 4 KB
  __shared__ float sRedD[8][M_TILE];
  __shared__ int   sRedI[8][M_TILE];
  __shared__ int   sIdx[M_TILE];
  __shared__ float sLoss[8];

  const int t = threadIdx.x;
  const int w = t >> 5;          // wave id 0..7
  const int l = t & 31;          // lane id
  const long m0 = (long)blockIdx.x * M_TILE;

  // --- load A tile coalesced: thread t = column, i = row ---
#pragma unroll
  for (int i = 0; i < M_TILE; ++i) {
    float v = x[(m0 + i) * D_DIM + t];
    sAf[i][t] = v;
    sAb[i][t] = (__bf16)v;
  }
#pragma unroll
  for (int i = 0; i < 4; ++i) sEn[t + i * 256] = enorm[t + i * 256];
  __syncthreads();
  if (t < M_TILE) {
    float s = 0.0f;
    for (int d = 0; d < D_DIM; ++d) { float v = sAf[t][d]; s += v * v; }
    sXn[t] = s;
  }
  __syncthreads();

  const int half = l >> 4;       // 0: lanes 0-15, 1: lanes 16-31
  const int lm   = l & 15;
  const int akb  = half * 8;     // A-fragment K chunk base (ISA 16-bit A layout)

  // --- preload both row-tiles' A fragments into registers (reused 8x) ---
  v16bf af0[8], af1[8];
#pragma unroll
  for (int ks = 0; ks < 8; ++ks) {
    const int d0 = ks * 32;
    v8bf a0lo = *(const v8bf*)&sAb[lm][d0 + akb];
    v8bf a0hi = *(const v8bf*)&sAb[lm][d0 + akb + 16];
    af0[ks] = __builtin_shufflevector(a0lo, a0hi,
               0, 1, 2, 3, 4, 5, 6, 7, 8, 9, 10, 11, 12, 13, 14, 15);
    v8bf a1lo = *(const v8bf*)&sAb[16 + lm][d0 + akb];
    v8bf a1hi = *(const v8bf*)&sAb[16 + lm][d0 + akb + 16];
    af1[ks] = __builtin_shufflevector(a1lo, a1hi,
               0, 1, 2, 3, 4, 5, 6, 7, 8, 9, 10, 11, 12, 13, 14, 15);
  }

  float bd0[8], bd1[8];
  int   bi0[8], bi1[8];
#pragma unroll
  for (int r = 0; r < 8; ++r) {
    bd0[r] = 3.0e38f; bi0[r] = 0;
    bd1[r] = 3.0e38f; bi1[r] = 0;
  }

  for (int ct8 = 0; ct8 < 8; ++ct8) {
    const int c0 = (ct8 * 8 + w) * 16;                 // first code of tile
    const __bf16* bp = eTb + (c0 + lm) * D_DIM + half * 8;
    if (ct8 < 7) __builtin_prefetch(bp + 128 * D_DIM, 0, 3);  // next tile's B

    v8f acc0 = {};
    v8f acc1 = {};
#pragma unroll
    for (int ks = 0; ks < 8; ++ks) {
      // B fragment: lane = code column, dims 8h..8h+7 and 16+8h..16+8h+7 (L2-hot),
      // shared by both row-tiles -> 2 WMMAs per 32B/lane of B traffic
      v8bf blo = *(const v8bf*)(bp + ks * 32);
      v8bf bhi = *(const v8bf*)(bp + ks * 32 + 16);
      v16bf b = __builtin_shufflevector(blo, bhi,
                 0, 1, 2, 3, 4, 5, 6, 7, 8, 9, 10, 11, 12, 13, 14, 15);
      acc0 = __builtin_amdgcn_wmma_f32_16x16x32_bf16(
          false, af0[ks], false, b, (short)0, acc0, false, false);
      acc1 = __builtin_amdgcn_wmma_f32_16x16x32_bf16(
          false, af1[ks], false, b, (short)0, acc1, false, false);
    }
    // dist = |x|^2 + |e|^2 - 2*sim ; running argmin per (row,lane-column)
#pragma unroll
    for (int r = 0; r < 8; ++r) {
      const int m = r + half * 8;
      const int code = c0 + lm;
      float en = sEn[code];
      float d0 = sXn[m] + en - 2.0f * acc0[r];
      if (d0 < bd0[r]) { bd0[r] = d0; bi0[r] = code; }
      float d1 = sXn[m + 16] + en - 2.0f * acc1[r];
      if (d1 < bd1[r]) { bd1[r] = d1; bi1[r] = code; }
    }
  }

  // --- reduce across the 16 columns (lanes within each half, wave32 shfl) ---
#pragma unroll
  for (int off = 8; off >= 1; off >>= 1) {
#pragma unroll
    for (int r = 0; r < 8; ++r) {
      float od = __shfl_xor(bd0[r], off, 32);
      int   oi = __shfl_xor(bi0[r], off, 32);
      if (od < bd0[r] || (od == bd0[r] && oi < bi0[r])) { bd0[r] = od; bi0[r] = oi; }
      od = __shfl_xor(bd1[r], off, 32);
      oi = __shfl_xor(bi1[r], off, 32);
      if (od < bd1[r] || (od == bd1[r] && oi < bi1[r])) { bd1[r] = od; bi1[r] = oi; }
    }
  }
  if (lm == 0) {
#pragma unroll
    for (int r = 0; r < 8; ++r) {
      const int m = r + half * 8;
      sRedD[w][m]      = bd0[r];
      sRedI[w][m]      = bi0[r];
      sRedD[w][m + 16] = bd1[r];
      sRedI[w][m + 16] = bi1[r];
    }
  }
  __syncthreads();

  // --- cross-wave argmin, first-index tie-break (matches jnp.argmin) ---
  if (t < M_TILE) {
    float best = sRedD[0][t];
    int   bidx = sRedI[0][t];
#pragma unroll
    for (int ww = 1; ww < 8; ++ww) {
      float dd = sRedD[ww][t];
      int   ii = sRedI[ww][t];
      if (dd < best || (dd == best && ii < bidx)) { best = dd; bidx = ii; }
    }
    sIdx[t] = bidx;
  }
  __syncthreads();

  // --- gather quantized rows (coalesced from f32-transposed codebook) + loss ---
  float total = 0.0f;
#pragma unroll
  for (int i = 0; i < M_TILE; ++i) {
    const int code = sIdx[i];
    float q  = eTf[code * D_DIM + t];
    float xv = sAf[i][t];
    float df = q - xv;
    total += df * df;
    out[(m0 + i) * D_DIM + t] = q;   // straight-through output == gathered code
  }
#pragma unroll
  for (int off = 16; off >= 1; off >>= 1) total += __shfl_xor(total, off, 32);
  if (l == 0) sLoss[w] = total;
  __syncthreads();
  if (t == 0) {
    float bsum = 0.0f;
#pragma unroll
    for (int ww = 0; ww < 8; ++ww) bsum += sLoss[ww];
    atomicAdd(lossptr, bsum * loss_scale);
  }
}

// ---------------------------------------------------------------------------
extern "C" void kernel_launch(void* const* d_in, const int* in_sizes, int n_in,
                              void* d_out, int out_size, void* d_ws, size_t ws_size,
                              hipStream_t stream) {
  const float* x   = (const float*)d_in[0];   // [32,32,32,256] f32
  const float* emb = (const float*)d_in[1];   // [256,1024] f32

  float* out = (float*)d_out;
  const int n_elems = out_size - 1;           // 8388608 quantized elements
  float* lossptr = out + n_elems;

  char* ws = (char*)d_ws;
  float*  eTf   = (float*)ws;                               // 1 MB  [K][D] f32
  __bf16* eTb   = (__bf16*)(ws + (1u << 20));               // 512KB [K][D] bf16
  float*  enorm = (float*)(ws + (1u << 20) + (1u << 19));   // 4 KB

  // loss = (beta + 1) * mean((q - x)^2) = 1.25 * sum / (N*D)
  const float loss_scale = 1.25f / (float)n_elems;

  vq_prep<<<K_CODES, 256, 0, stream>>>(emb, eTf, eTb, enorm, lossptr);
  vq_main<<<(n_elems / D_DIM) / M_TILE, 256, 0, stream>>>(
      x, eTf, eTb, enorm, out, lossptr, loss_scale);
}
